// MultiHeadAttention_78546361909880
// MI455X (gfx1250) — matrix operand order repaired and checked by reading, not compiled, not run on hardware
//
#include <hip/hip_runtime.h>
#include <stdint.h>

#define BB 4
#define SS 1024
#define DD 1024
#define HH 16
#define DEPTH 64
#define MM (BB*SS)   // 4096

typedef __attribute__((ext_vector_type(16))) __bf16 v16bf;
typedef __attribute__((ext_vector_type(8)))  float  v8f;
typedef __attribute__((ext_vector_type(4)))  int    v4i;

union FragBF { v16bf v; uint4 u[2]; };
union AccF   { v8f   v; float f[8]; };

__device__ __forceinline__ uint16_t f2bf(float x) {
  union { float f; uint32_t u; } a; a.f = x;
  uint32_t u = a.u;
  uint32_t r = u + 0x7FFFu + ((u >> 16) & 1u);   // round-to-nearest-even
  return (uint16_t)(r >> 16);
}
__device__ __forceinline__ uint32_t f2bf_pk(float lo, float hi) {
  return (uint32_t)f2bf(lo) | ((uint32_t)f2bf(hi) << 16);
}

// ---- gfx1250 async global->LDS copy (ASYNCcnt path), guarded fallback ------
#if __has_builtin(__builtin_amdgcn_global_load_async_to_lds_b128)
#define HAVE_ASYNC_LDS 1
typedef __attribute__((address_space(1))) v4i gv4i;   // global int4
typedef __attribute__((address_space(3))) v4i sv4i;   // LDS int4
#else
#define HAVE_ASYNC_LDS 0
#endif

__device__ __forceinline__ void copy16_g2l(const uint16_t* g, uint16_t* l) {
#if HAVE_ASYNC_LDS
  __builtin_amdgcn_global_load_async_to_lds_b128((gv4i*)g, (sv4i*)l, 0, 0);
#else
  *(uint4*)l = *(const uint4*)g;
#endif
}
__device__ __forceinline__ void async_fence() {
#if HAVE_ASYNC_LDS
# if __has_builtin(__builtin_amdgcn_s_wait_asynccnt)
  __builtin_amdgcn_s_wait_asynccnt(0);
# else
  asm volatile("s_wait_asynccnt 0" ::: "memory");
# endif
#endif
}

// ---------------------------------------------------------------------------
// Projection GEMM: out[m,n] = sum_k A[m,k] * W[n,k] + bias[n]
// MODE 0: A=q fp32 -> bf16 Qh scratch (scaled by 1/8)
// MODE 1: A=k fp32 -> bf16 Kh scratch + fp32 present_k
// MODE 2: A=v fp32 -> bf16 Vh scratch + fp32 present_v
// MODE 3: A=merged bf16 -> fp32 final output
// Block tile 128x128, 256 thr (8 waves), wave tile 64x32, K-step 32.
// ---------------------------------------------------------------------------
#define LDA_P 40   // padded LDS row stride (bf16 elems): 80 B, multiple of 16 B

template<int MODE>
__global__ __launch_bounds__(256) void gemm_proj(
    const float*    __restrict__ Af,
    const uint16_t* __restrict__ Ab,
    const float*    __restrict__ W,
    const float*    __restrict__ bias,
    uint16_t*       __restrict__ out_bf,
    float*          __restrict__ out_f)
{
  __shared__ uint16_t sA[128 * LDA_P];
  __shared__ uint16_t sB[128 * LDA_P];

  const int tid  = threadIdx.x;
  const int wave = tid >> 5;
  const int lane = tid & 31;
  const int half = lane >> 4;
  const int l16  = lane & 15;
  const int m0   = blockIdx.x * 128;
  const int n0   = blockIdx.y * 128;
  const int wm   = (wave & 1) * 64;
  const int wn   = (wave >> 1) * 32;

  AccF acc[4][2];
  #pragma unroll
  for (int i = 0; i < 4; ++i)
    #pragma unroll
    for (int j = 0; j < 2; ++j)
      #pragma unroll
      for (int e = 0; e < 8; ++e) acc[i][j].f[e] = 0.0f;

  for (int k0 = 0; k0 < DD; k0 += 32) {
    // cooperative stage: A tile 128x32, B tile 128x32 (fp32 -> bf16, packed b64 stores)
    #pragma unroll
    for (int it = 0; it < 4; ++it) {
      int slot = tid + it * 256;       // 0..1023 = 128 rows x 8 quad-chunks
      int row  = slot >> 3;
      int kc   = (slot & 7) * 4;
      if (MODE < 3) {
        const float* ap = Af + (size_t)(m0 + row) * DD + k0 + kc;
        const float4 a4 = *(const float4*)ap;
        uint2 pa; pa.x = f2bf_pk(a4.x, a4.y); pa.y = f2bf_pk(a4.z, a4.w);
        *(uint2*)&sA[row * LDA_P + kc] = pa;
        if (k0 + 32 < DD) __builtin_prefetch(ap + 32, 0, 1);
      } else {
        const uint16_t* ap = Ab + (size_t)(m0 + row) * DD + k0 + kc;
        *(uint2*)&sA[row * LDA_P + kc] = *(const uint2*)ap;
        if (k0 + 32 < DD) __builtin_prefetch(ap + 32, 0, 1);
      }
      const float* bp = W + (size_t)(n0 + row) * DD + k0 + kc;
      const float4 b4 = *(const float4*)bp;
      uint2 pb; pb.x = f2bf_pk(b4.x, b4.y); pb.y = f2bf_pk(b4.z, b4.w);
      *(uint2*)&sB[row * LDA_P + kc] = pb;
      if (k0 + 32 < DD) __builtin_prefetch(bp + 32, 0, 1);
    }
    __syncthreads();

    FragBF afr[4], bfr[2];
    #pragma unroll
    for (int tm = 0; tm < 4; ++tm) {
      const uint16_t* src = &sA[(wm + tm * 16 + l16) * LDA_P];
      afr[tm].u[0] = *(const uint4*)(src + half * 8);        // K 0..7  / 8..15
      afr[tm].u[1] = *(const uint4*)(src + 16 + half * 8);   // K 16..23 / 24..31
    }
    #pragma unroll
    for (int tn = 0; tn < 2; ++tn) {
      const uint16_t* src = &sB[(wn + tn * 16 + l16) * LDA_P];
      bfr[tn].u[0] = *(const uint4*)(src + half * 8);        // K 0..7  / 8..15
      bfr[tn].u[1] = *(const uint4*)(src + 16 + half * 8);   // K 16..23 / 24..31
    }
    #pragma unroll
    for (int tm = 0; tm < 4; ++tm)
      #pragma unroll
      for (int tn = 0; tn < 2; ++tn)
        acc[tm][tn].v = __builtin_amdgcn_wmma_f32_16x16x32_bf16(
            false, afr[tm].v, false, bfr[tn].v, (short)0, acc[tm][tn].v, false, false);
    __syncthreads();
  }

  // epilogue
  #pragma unroll
  for (int tm = 0; tm < 4; ++tm) {
    #pragma unroll
    for (int tn = 0; tn < 2; ++tn) {
      const int colb = n0 + wn + tn * 16 + l16;
      const float bv = bias[colb];
      #pragma unroll
      for (int r = 0; r < 8; ++r) {
        const int row = m0 + wm + tm * 16 + half * 8 + r;
        const float val = acc[tm][tn].f[r] + bv;
        if (MODE == 3) {
          out_f[(size_t)row * DD + colb] = val;
        } else {
          const int b = row >> 10;         // / S
          const int s = row & (SS - 1);
          const int h = colb >> 6;         // / DEPTH
          const int d = colb & (DEPTH - 1);
          const size_t idx = (((size_t)b * HH + h) * SS + s) * DEPTH + d;
          out_bf[idx] = f2bf((MODE == 0) ? val * 0.125f : val);
          if (MODE == 1 || MODE == 2) out_f[idx] = val;   // present half
        }
      }
    }
  }
}

// ---------------------------------------------------------------------------
// Flash attention: one block (4 waves, 128 thr) per (b, h, 64-query rows).
// Streams causal 64-key blocks; QK^T and PV via bf16 WMMA, fp32 online softmax.
// Q/K tiles staged with gfx1250 async global->LDS copies when available.
// ---------------------------------------------------------------------------
#define LQ 72   // padded LDS row stride (bf16): 144 B, multiple of 16 B

__global__ __launch_bounds__(128) void attn_kernel(
    const uint16_t* __restrict__ Qh,   // [B,H,S,64] bf16, pre-scaled by 1/8
    const uint16_t* __restrict__ Kh,
    const uint16_t* __restrict__ Vh,
    const float*    __restrict__ attn_mask,  // [B,1,1,S]
    const float*    __restrict__ head_mask,  // [1,H,1,1]
    uint16_t*       __restrict__ merged)     // [B,S,D] bf16
{
  __shared__ uint16_t sQ [64 * LQ];
  __shared__ uint16_t sK [64 * LQ];
  __shared__ uint16_t sVt[64 * LQ];   // transposed: [d][key]
  __shared__ uint16_t sP [64 * LQ];

  const int bh   = blockIdx.x;
  const int b    = bh >> 4;
  const int h    = bh & 15;
  const int i0   = blockIdx.y * 64;
  const int tid  = threadIdx.x;
  const int wave = tid >> 5;
  const int lane = tid & 31;
  const int half = lane >> 4;
  const int l16  = lane & 15;

  const size_t base = (((size_t)b * HH + h) * SS) * DEPTH;

  // stage Q tile 64x64 (async when available)
  #pragma unroll
  for (int it = 0; it < 4; ++it) {
    int slot = tid + it * 128;         // 64 rows x 8 oct-chunks
    int row  = slot >> 3;
    int kc   = (slot & 7) * 8;
    copy16_g2l(Qh + base + (size_t)(i0 + row) * DEPTH + kc, &sQ[row * LQ + kc]);
  }

  float mrow[8], lrow[8];
  #pragma unroll
  for (int r = 0; r < 8; ++r) { mrow[r] = -1e30f; lrow[r] = 0.0f; }
  AccF o[4];
  #pragma unroll
  for (int t = 0; t < 4; ++t)
    #pragma unroll
    for (int e = 0; e < 8; ++e) o[t].f[e] = 0.0f;

  const int jmax = i0 >> 6;            // causal block limit
  for (int jb = 0; jb <= jmax; ++jb) {
    const int j0 = jb * 64;
    __syncthreads();
    // stage K tile (async) and V^T tile (manual transpose)
    #pragma unroll
    for (int it = 0; it < 4; ++it) {
      int slot = tid + it * 128;
      int row  = slot >> 3;
      int kc   = (slot & 7) * 8;
      copy16_g2l(Kh + base + (size_t)(j0 + row) * DEPTH + kc, &sK[row * LQ + kc]);
      uint4 u = *(const uint4*)(Vh + base + (size_t)(j0 + row) * DEPTH + kc);
      const uint16_t* e = (const uint16_t*)&u;
      #pragma unroll
      for (int q = 0; q < 8; ++q) sVt[(kc + q) * LQ + row] = e[q];
    }
    async_fence();
    __syncthreads();

    // S = Q K^T  (wave rows wave*16..+15 x 64 key cols)
    AccF sc[4];
    #pragma unroll
    for (int t = 0; t < 4; ++t)
      #pragma unroll
      for (int e = 0; e < 8; ++e) sc[t].f[e] = 0.0f;
    #pragma unroll
    for (int ks = 0; ks < 2; ++ks) {
      FragBF aq;
      const uint16_t* qsrc = &sQ[(wave * 16 + l16) * LQ + ks * 32];
      aq.u[0] = *(const uint4*)(qsrc + half * 8);
      aq.u[1] = *(const uint4*)(qsrc + 16 + half * 8);
      #pragma unroll
      for (int t = 0; t < 4; ++t) {
        FragBF bk;
        const uint16_t* ksrc = &sK[(t * 16 + l16) * LQ + ks * 32];
        bk.u[0] = *(const uint4*)(ksrc + half * 8);
        bk.u[1] = *(const uint4*)(ksrc + 16 + half * 8);
        sc[t].v = __builtin_amdgcn_wmma_f32_16x16x32_bf16(
            false, aq.v, false, bk.v, (short)0, sc[t].v, false, false);
      }
    }

    // masks + online softmax
    const int row0 = i0 + wave * 16 + half * 8;
    #pragma unroll
    for (int t = 0; t < 4; ++t) {
      const int colg = j0 + t * 16 + l16;
      const float am = attn_mask[(size_t)b * SS + colg];
      #pragma unroll
      for (int r = 0; r < 8; ++r) {
        float vv = sc[t].f[r] + am;
        if (jb == jmax && colg > (row0 + r)) vv = -1e30f;   // causal
        sc[t].f[r] = vv;
      }
    }
    #pragma unroll
    for (int r = 0; r < 8; ++r) {
      float bm = -1e30f;
      #pragma unroll
      for (int t = 0; t < 4; ++t) bm = fmaxf(bm, sc[t].f[r]);
      #pragma unroll
      for (int off = 1; off < 16; off <<= 1) bm = fmaxf(bm, __shfl_xor(bm, off, 32));
      const float mnew  = fmaxf(mrow[r], bm);
      const float alpha = __expf(mrow[r] - mnew);
      mrow[r] = mnew;
      float psum = 0.0f;
      #pragma unroll
      for (int t = 0; t < 4; ++t) {
        const float p = __expf(sc[t].f[r] - mnew);
        sc[t].f[r] = p;
        psum += p;
      }
      #pragma unroll
      for (int off = 1; off < 16; off <<= 1) psum += __shfl_xor(psum, off, 32);
      lrow[r] = lrow[r] * alpha + psum;
      #pragma unroll
      for (int t = 0; t < 4; ++t) o[t].f[r] *= alpha;
    }

    // P -> LDS (wave-private rows), then O += P * V
    #pragma unroll
    for (int t = 0; t < 4; ++t)
      #pragma unroll
      for (int r = 0; r < 8; ++r)
        sP[(wave * 16 + half * 8 + r) * LQ + t * 16 + l16] = f2bf(sc[t].f[r]);
    asm volatile("s_wait_dscnt 0" ::: "memory");   // order wave-private LDS store->load

    #pragma unroll
    for (int ks = 0; ks < 2; ++ks) {
      FragBF ap;
      const uint16_t* psrc = &sP[(wave * 16 + l16) * LQ + ks * 32];
      ap.u[0] = *(const uint4*)(psrc + half * 8);
      ap.u[1] = *(const uint4*)(psrc + 16 + half * 8);
      #pragma unroll
      for (int t = 0; t < 4; ++t) {
        FragBF bv;
        const uint16_t* vsrc = &sVt[(t * 16 + l16) * LQ + ks * 32];
        bv.u[0] = *(const uint4*)(vsrc + half * 8);
        bv.u[1] = *(const uint4*)(vsrc + 16 + half * 8);
        o[t].v = __builtin_amdgcn_wmma_f32_16x16x32_bf16(
            false, ap.v, false, bv.v, (short)0, o[t].v, false, false);
      }
    }
  }

  // normalize, apply head_mask, write merged bf16 [B,S,D]
  const float hm = head_mask[h];
  #pragma unroll
  for (int t = 0; t < 4; ++t) {
    #pragma unroll
    for (int r = 0; r < 8; ++r) {
      const float vv   = o[t].f[r] * hm / lrow[r];
      const int   srow = i0 + wave * 16 + half * 8 + r;
      const int   dcol = t * 16 + l16;
      merged[((size_t)b * SS + srow) * DD + h * DEPTH + dcol] = f2bf(vv);
    }
  }
}

// ---------------------------------------------------------------------------
extern "C" void kernel_launch(void* const* d_in, const int* in_sizes, int n_in,
                              void* d_out, int out_size, void* d_ws, size_t ws_size,
                              hipStream_t stream) {
  (void)in_sizes; (void)n_in; (void)out_size; (void)ws_size;
  const float* q         = (const float*)d_in[0];
  const float* k         = (const float*)d_in[1];
  const float* v         = (const float*)d_in[2];
  /* d_in[3] = causal mask, recomputed analytically in-kernel */
  const float* attn_mask = (const float*)d_in[4];
  const float* head_mask = (const float*)d_in[5];
  const float* Wq_w = (const float*)d_in[6];
  const float* Wq_b = (const float*)d_in[7];
  const float* Wk_w = (const float*)d_in[8];
  const float* Wk_b = (const float*)d_in[9];
  const float* Wv_w = (const float*)d_in[10];
  const float* Wv_b = (const float*)d_in[11];
  const float* dense_w = (const float*)d_in[12];
  const float* dense_b = (const float*)d_in[13];

  float* out       = (float*)d_out;                       // [B,S,D]
  float* present_k = out + (size_t)BB * SS * DD;          // [B,H,S,64]
  float* present_v = present_k + (size_t)BB * HH * SS * DEPTH;

  const size_t hd = (size_t)BB * HH * SS * DEPTH;         // 4M elems
  uint16_t* Qh     = (uint16_t*)d_ws;
  uint16_t* Kh     = Qh + hd;
  uint16_t* Vh     = Kh + hd;
  uint16_t* merged = Vh + hd;

  dim3 gblk(32, 8);
  gemm_proj<0><<<gblk, 256, 0, stream>>>(q, nullptr, Wq_w, Wq_b, Qh, nullptr);
  gemm_proj<1><<<gblk, 256, 0, stream>>>(k, nullptr, Wk_w, Wk_b, Kh, present_k);
  gemm_proj<2><<<gblk, 256, 0, stream>>>(v, nullptr, Wv_w, Wv_b, Vh, present_v);
  attn_kernel<<<dim3(64, 16), 128, 0, stream>>>(Qh, Kh, Vh, attn_mask, head_mask, merged);
  gemm_proj<3><<<gblk, 256, 0, stream>>>(nullptr, merged, dense_w, dense_b, nullptr, out);
}
